// VectorQuantizer_7404523619098
// MI455X (gfx1250) — matrix operand order repaired and checked by reading, not compiled, not run on hardware
//
#include <hip/hip_runtime.h>
#include <hip/hip_bf16.h>

typedef __attribute__((ext_vector_type(16))) __bf16 v16bf;
typedef __attribute__((ext_vector_type(8)))  float  v8f;

constexpr int Bn = 32768;   // rows of x
constexpr int Kn = 8192;    // codebook entries
constexpr int Dn = 256;     // embedding dim
constexpr int LDSS = 264;   // padded bf16 row stride in LDS: 528B = 33*16B, banks spread 4/row
constexpr int NTILES = Kn / 16;  // 512 codebook N-tiles

#if __has_builtin(__builtin_amdgcn_tensor_load_to_lds)
#define VQ_USE_TDM 1
typedef __attribute__((ext_vector_type(4))) unsigned int v4u;
typedef __attribute__((ext_vector_type(8))) int          v8i_;
typedef __attribute__((ext_vector_type(4))) int          v4i_;

// Build a 2-D TDM descriptor for one 16x256 bf16 codebook tile -> padded LDS
// tile (row stride 528B via pad_enable: +4 DWORDs every 128 DWORDs).
__device__ __forceinline__ void tdm_load_tile(const __bf16* gsrc, unsigned lds_off) {
  unsigned long long ga = (unsigned long long)gsrc;
  v4u g0;
  g0[0] = 1u;                                     // count=1, user descriptor
  g0[1] = lds_off;                                // D#.lds_addr (bytes)
  g0[2] = (unsigned)ga;                           // global_addr[31:0]
  g0[3] = (unsigned)((ga >> 32) & 0x1ffffffull)   // global_addr[56:32]
        | (2u << 30);                             // type = 2 ("image")
  v8i_ g1;
  g1[0] = (int)((1u << 16)      // data_size = 1 -> 2-byte elements (bf16)
        | (1u << 20)            // pad_enable (LDS dest padding)
        | (6u << 22)            // pad_interval: 128 DWORDs (= one 512B row)
        | (3u << 25));          // pad_amount: 4 DWORDs (16B) -> 528B LDS row stride
  g1[1] = (int)(((unsigned)(Dn & 0xffff)) << 16);                 // tensor_dim0 lo16
  g1[2] = (int)(((unsigned)Dn >> 16) | (((unsigned)Kn & 0xffffu) << 16)); // dim0 hi | dim1 lo
  g1[3] = (int)(((unsigned)Kn >> 16) | ((unsigned)Dn << 16));     // dim1 hi | tile_dim0=256
  g1[4] = 16;                                                     // tile_dim1=16, tile_dim2=0
  g1[5] = Dn;                                                     // tensor_dim0_stride lo32
  g1[6] = 0;                                                      // stride0 hi | stride1 lo
  g1[7] = 0;                                                      // stride1 hi
  v4i_ z = {0, 0, 0, 0};
#if defined(__clang_major__) && (__clang_major__ >= 23)
  v8i_ z8 = {0, 0, 0, 0, 0, 0, 0, 0};
  __builtin_amdgcn_tensor_load_to_lds(g0, g1, z, z, z8, 0);
#else
  __builtin_amdgcn_tensor_load_to_lds(g0, g1, z, z, 0);
#endif
}
#endif

// ---------------------------------------------------------------------------
// Kernel 1: codebook f32 -> bf16 (row major) and C2[k] = ||codebook[k]||^2
// One wave32 per codebook row; 8 floats per lane.
// ---------------------------------------------------------------------------
__global__ __launch_bounds__(256) void vq_prep_kernel(
    const float* __restrict__ cb, __bf16* __restrict__ cbh,
    float* __restrict__ c2) {
  const int tid  = threadIdx.x;
  const int lane = tid & 31;
  const int wave = tid >> 5;
  const int row  = blockIdx.x * 8 + wave;

  const float* src = cb + (size_t)row * Dn + lane * 8;
  float4 f0 = ((const float4*)src)[0];
  float4 f1 = ((const float4*)src)[1];

  union { __bf16 h[8]; uint4 u; } pk;
  pk.h[0] = (__bf16)f0.x; pk.h[1] = (__bf16)f0.y;
  pk.h[2] = (__bf16)f0.z; pk.h[3] = (__bf16)f0.w;
  pk.h[4] = (__bf16)f1.x; pk.h[5] = (__bf16)f1.y;
  pk.h[6] = (__bf16)f1.z; pk.h[7] = (__bf16)f1.w;
  *(uint4*)(cbh + (size_t)row * Dn + lane * 8) = pk.u;

  float s = f0.x*f0.x + f0.y*f0.y + f0.z*f0.z + f0.w*f0.w
          + f1.x*f1.x + f1.y*f1.y + f1.z*f1.z + f1.w*f1.w;
  #pragma unroll
  for (int off = 16; off >= 1; off >>= 1)
    s += __shfl_xor(s, off, 32);
  if (lane == 0) c2[row] = s;
}

// ---------------------------------------------------------------------------
// Kernel 2: per-row argmin over K of (C2[n] - 2 x.c_n) via bf16 WMMA, then
// gather the winning f32 codebook row. Block = 128 threads = 4 waves; each
// wave owns M=32 rows of x (two 16-row C tiles -> every LDS B-fragment feeds
// two WMMAs). Codebook streamed through double-buffered LDS tiles, staged by
// the Tensor Data Mover when available. B fragments are software-pipelined
// (3 buffers, 2 chunks ahead) so WMMA never waits on a just-issued ds_load.
// ---------------------------------------------------------------------------
__global__ __launch_bounds__(128) void vq_main_kernel(
    const float* __restrict__ x, const float* __restrict__ cb,
    const __bf16* __restrict__ cbh, const float* __restrict__ c2g,
    float* __restrict__ out) {
  __shared__ ushort scb[2][16 * LDSS];   // 2 x 8448 B padded codebook tiles (bf16)
  __shared__ float  sc2[Kn];             // 32 KB of ||c||^2

  const int tid  = threadIdx.x;
  const int lane = tid & 31;
  const int wave = tid >> 5;            // 0..3
  const int s    = lane >> 4;           // lane half (ISA k-segment select)
  const int nl   = lane & 15;           // row-within-half / N column

  // Stage all C2 into LDS once.
  for (int i = tid; i < Kn; i += 128) sc2[i] = c2g[i];

  // ---- Build this wave's A fragments: 32 x 256 of x, bf16, 8 chunks of K=32.
  // ISA 16-bit A layout: lane half s holds k = 32c+8s+[0..8) and 32c+16+8s+[0..8).
  const int mbase = blockIdx.x * 128 + wave * 32;
  v16bf afrag[2][8];
  #pragma unroll
  for (int h = 0; h < 2; ++h) {
    const float* xr = x + (size_t)(mbase + h * 16 + nl) * Dn;
    #pragma unroll
    for (int c = 0; c < 8; ++c) {
      const int k1 = 32 * c + 8 * s;
      float4 f0 = *(const float4*)(xr + k1);
      float4 f1 = *(const float4*)(xr + k1 + 4);
      float4 f2 = *(const float4*)(xr + k1 + 16);
      float4 f3 = *(const float4*)(xr + k1 + 20);
      v16bf a;
      a[0]  = (__bf16)f0.x; a[1]  = (__bf16)f0.y; a[2]  = (__bf16)f0.z; a[3]  = (__bf16)f0.w;
      a[4]  = (__bf16)f1.x; a[5]  = (__bf16)f1.y; a[6]  = (__bf16)f1.z; a[7]  = (__bf16)f1.w;
      a[8]  = (__bf16)f2.x; a[9]  = (__bf16)f2.y; a[10] = (__bf16)f2.z; a[11] = (__bf16)f2.w;
      a[12] = (__bf16)f3.x; a[13] = (__bf16)f3.y; a[14] = (__bf16)f3.z; a[15] = (__bf16)f3.w;
      afrag[h][c] = a;
    }
  }

#if defined(VQ_USE_TDM)
  // Async DMA the first tile into buffer 0.
  if (wave == 0) {
    tdm_load_tile(cbh, (unsigned)(size_t)&scb[0][0]);
    __builtin_amdgcn_s_wait_tensorcnt((short)0);
  }
#else
  // Manual staging: thread -> (row, 64B chunk). 16 rows x 512B = 128 x 64B.
  const int srow = tid >> 3;          // 0..15 tile row
  const int scol = (tid & 7) * 4;     // 16B segment base, 0..28
  {
    const __bf16* src = cbh + (size_t)srow * Dn + scol * 8;
    ushort* dst = &scb[0][srow * LDSS + scol * 8];
    #pragma unroll
    for (int j = 0; j < 4; ++j)
      ((uint4*)dst)[j] = ((const uint4*)src)[j];
  }
#endif
  __syncthreads();

  float minv[2][8]; int mini[2][8];
  #pragma unroll
  for (int h = 0; h < 2; ++h)
    #pragma unroll
    for (int r = 0; r < 8; ++r) { minv[h][r] = 3.4028235e38f; mini[h][r] = 0; }

  union BF { uint4 u[2]; v16bf v; };

  for (int t = 0; t < NTILES; ++t) {
    const int buf = t & 1;
    const bool have_next = (t + 1) < NTILES;

#if defined(VQ_USE_TDM)
    if (have_next && wave == 0)
      tdm_load_tile(cbh + (size_t)(t + 1) * 16 * Dn,
                    (unsigned)(size_t)&scb[buf ^ 1][0]);
#else
    uint4 nv[4];
    if (have_next) {
      const __bf16* src = cbh + (size_t)((t + 1) * 16 + srow) * Dn + scol * 8;
      #pragma unroll
      for (int j = 0; j < 4; ++j) nv[j] = ((const uint4*)src)[j];
      if ((t + 2) < NTILES)
        __builtin_prefetch(cbh + (size_t)((t + 2) * 16 + srow) * Dn, 0, 3);
    }
#endif

    // Per-tile score column and its ||c||^2 (hoisted before the B-load chain).
    const int   nidx = t * 16 + nl;
    const float c2v  = sc2[nidx];

    // Two 16x16 dot-product tiles sharing every B fragment (K = 8 x 32 = 256).
    // 3-buffer rotation, loads issued 2 chunks ahead of their WMMA consumers.
    const ushort* bp = &scb[buf][nl * LDSS];
    BF bf[3];
    #define VQ_LDB(c, d)                                                \
      do {                                                              \
        const uint4* q = (const uint4*)(bp + 32 * (c) + 8 * s);         \
        (d).u[0] = q[0];                                                \
        (d).u[1] = q[2];                                                \
      } while (0)

    VQ_LDB(0, bf[0]);
    VQ_LDB(1, bf[1]);
    v8f acc0 = {0.f, 0.f, 0.f, 0.f, 0.f, 0.f, 0.f, 0.f};
    v8f acc1 = {0.f, 0.f, 0.f, 0.f, 0.f, 0.f, 0.f, 0.f};
    #pragma unroll
    for (int c = 0; c < 8; ++c) {
      if (c + 2 < 8) VQ_LDB(c + 2, bf[(c + 2) % 3]);
      acc0 = __builtin_amdgcn_wmma_f32_16x16x32_bf16(
          false, afrag[0][c], false, bf[c % 3].v, (short)0, acc0, false, false);
      acc1 = __builtin_amdgcn_wmma_f32_16x16x32_bf16(
          false, afrag[1][c], false, bf[c % 3].v, (short)0, acc1, false, false);
    }
    #undef VQ_LDB

    // Running argmin update: score = C2[n] - 2*dot  (strict '<' keeps first idx)
    #pragma unroll
    for (int r = 0; r < 8; ++r) {
      float s0 = fmaf(-2.0f, acc0[r], c2v);
      float s1 = fmaf(-2.0f, acc1[r], c2v);
      if (s0 < minv[0][r]) { minv[0][r] = s0; mini[0][r] = nidx; }
      if (s1 < minv[1][r]) { minv[1][r] = s1; mini[1][r] = nidx; }
    }

#if defined(VQ_USE_TDM)
    if (have_next && wave == 0)
      __builtin_amdgcn_s_wait_tensorcnt((short)0);
#else
    if (have_next) {
      ushort* dst = &scb[buf ^ 1][srow * LDSS + scol * 8];
      #pragma unroll
      for (int j = 0; j < 4; ++j) ((uint4*)dst)[j] = nv[j];
    }
#endif
    __syncthreads();
  }

  // Cross-lane min+index reduction within each 16-lane half (C layout rows),
  // then gather the winning f32 codebook rows (32 rows x 256 floats per wave).
  #pragma unroll
  for (int h = 0; h < 2; ++h) {
    #pragma unroll
    for (int off = 8; off >= 1; off >>= 1) {
      #pragma unroll
      for (int r = 0; r < 8; ++r) {
        float ov = __shfl_xor(minv[h][r], off, 32);
        int   oi = __shfl_xor(mini[h][r], off, 32);
        if (ov < minv[h][r] || (ov == minv[h][r] && oi < mini[h][r])) {
          minv[h][r] = ov; mini[h][r] = oi;
        }
      }
    }
    #pragma unroll
    for (int r = 0; r < 16; ++r) {
      const int idx = __shfl(mini[h][r & 7], (r < 8) ? 0 : 16, 32);
      const float4* sp = (const float4*)(cb + (size_t)idx * Dn) + lane * 2;
      float4* dp = (float4*)(out + (size_t)(mbase + h * 16 + r) * Dn) + lane * 2;
      dp[0] = sp[0];
      dp[1] = sp[1];
    }
  }
}

// ---------------------------------------------------------------------------
extern "C" void kernel_launch(void* const* d_in, const int* in_sizes, int n_in,
                              void* d_out, int out_size, void* d_ws, size_t ws_size,
                              hipStream_t stream) {
  const float* x  = (const float*)d_in[0];   // [B, D] f32
  const float* cb = (const float*)d_in[1];   // [K, D] f32
  float* out = (float*)d_out;                // [B, D] f32

  // Workspace layout: [0, 4MB) codebook bf16, then 32KB of C2.
  __bf16* cbh = (__bf16*)d_ws;
  float*  c2  = (float*)((char*)d_ws + (size_t)Kn * Dn * sizeof(__bf16));

  vq_prep_kernel<<<Kn / 8, 256, 0, stream>>>(cb, cbh, c2);
  vq_main_kernel<<<Bn / 128, 128, 0, stream>>>(x, cb, cbh, c2, out);
}
